// LSTM_46651934769186
// MI455X (gfx1250) — matrix operand order repaired and checked by reading, not compiled, not run on hardware
//
#include <hip/hip_runtime.h>
#include <hip/hip_bf16.h>

// ---------------- problem constants ----------------
#define H      265
#define INS    19
#define BATCH  512
#define TSTEPS 256
#define CHPAD  272      // gate chunk padded to 17 * 16
#define NTCH   17       // N-tiles per gate chunk
#define KTH    9        // k-tiles covering H padded to 288
#define HPAD   288      // padded K for h
#define NTILES (4*NTCH) // 68 total N tiles

typedef __attribute__((ext_vector_type(16))) __bf16 v16bf;
typedef __attribute__((ext_vector_type(8)))  float  v8f;

union BV { uint4 q[2]; v16bf v; };

__device__ __forceinline__ unsigned f2bf(float f) {
  unsigned u = __builtin_bit_cast(unsigned, f);
  u += 0x7fffu + ((u >> 16) & 1u);      // round-to-nearest-even
  return u >> 16;
}
__device__ __forceinline__ float sigf(float x)     { return 1.0f / (1.0f + __expf(-x)); }
__device__ __forceinline__ float tanhfast(float x) { return 1.0f - 2.0f / (__expf(2.0f * x) + 1.0f); }

__device__ __forceinline__ v8f wmma_bf16(v16bf a, v16bf b, v8f c) {
  return __builtin_amdgcn_wmma_f32_16x16x32_bf16(false, a, false, b, (short)0, c, false, false);
}

// A-tile (16x32 bf16) from LDS, per documented 16-bit A layout:
// lanes<16: VGPR0-3 = K0..7, VGPR4-7 = K16..23 ; lanes>=16: +8
__device__ __forceinline__ v16bf load_a(const unsigned short* base, int pitch, int lane, int kt) {
  int M   = lane & 15;
  int klo = (lane & 16) ? 8 : 0;
  BV u;
  u.q[0] = *(const uint4*)(base + M * pitch + kt * 32 + klo);       // ds_load_b128
  u.q[1] = *(const uint4*)(base + M * pitch + kt * 32 + 16 + klo);  // ds_load_b128
  return u.v;
}

// B-tile (32x16 bf16) from pre-packed global buffer: 32 contiguous bytes per lane
__device__ __forceinline__ v16bf load_b(const unsigned int* pk, int tile, int ktiles, int kt, int lane) {
  const uint4* p = (const uint4*)(pk + (((size_t)tile * ktiles + kt) * 32 + lane) * 8);
  BV u;
  u.q[0] = p[0];   // global_load_b128
  u.q[1] = p[1];   // global_load_b128
  return u.v;
}

// 9-k-tile GEMM accumulation for one chunk-local N tile m.
// unroll-3 gives the scheduler a window to keep ~24 global b128 loads in
// flight under the WMMA groups without manual rotation copies.
__device__ __forceinline__ void gemm_kloop(const unsigned short* hbase,
                                           const unsigned int* __restrict__ pk,
                                           int m, int lane, v8f (&acc)[4]) {
#pragma unroll 3
  for (int kt = 0; kt < KTH; kt++) {
    v16bf a = load_a(hbase, HPAD, lane, kt);
    acc[0] = wmma_bf16(a, load_b(pk, 0 * NTCH + m, KTH, kt, lane), acc[0]);
    acc[1] = wmma_bf16(a, load_b(pk, 1 * NTCH + m, KTH, kt, lane), acc[1]);
    acc[2] = wmma_bf16(a, load_b(pk, 2 * NTCH + m, KTH, kt, lane), acc[2]);
    acc[3] = wmma_bf16(a, load_b(pk, 3 * NTCH + m, KTH, kt, lane), acc[3]);
  }
}

// ---------------- weight pre-pack into WMMA-native layout ----------------
__global__ void pack_w(const float* __restrict__ w, int Kreal, int ktiles,
                       unsigned int* __restrict__ dst) {
  int blk  = blockIdx.x;                 // tile*ktiles + kt
  int tile = blk / ktiles;
  int kt   = blk - tile * ktiles;
  int tid  = threadIdx.x;
  int lane = tid >> 3, r = tid & 7;
  int chunk = tile / NTCH, mm = tile - chunk * NTCH;
  int nl = mm * 16 + (lane & 15);
  int n  = chunk * H + nl;
  int k0 = kt * 32 + ((lane & 16) ? 8 : 0) + 2 * r + ((r >= 4) ? 8 : 0);
  unsigned lo = 0, hc = 0;
  if (nl < H) {
    if (k0     < Kreal) lo = f2bf(w[(size_t)n * Kreal + k0]);
    if (k0 + 1 < Kreal) hc = f2bf(w[(size_t)n * Kreal + k0 + 1]);
  }
  dst[(size_t)blk * 256 + lane * 8 + r] = lo | (hc << 16);
}

__global__ void pack_bias(const float* __restrict__ bx, const float* __restrict__ bh,
                          float* __restrict__ dst) {
  int i = blockIdx.x * blockDim.x + threadIdx.x;
  if (i >= 4 * CHPAD) return;
  int chunk = i / CHPAD, j = i - chunk * CHPAD;
  dst[i] = (j < H) ? (bx[chunk * H + j] + bh[chunk * H + j]) : 0.0f;
}

// ---------------- persistent 2-layer LSTM, one 16-row batch tile per WG ----------------
// 8 waves; wave w owns chunk-local N tiles m = w, w+8 (and 16 for w==0).
// h0/h1/x are ping-pong LDS buffers -> only 2 barriers per time step.
__global__ __launch_bounds__(256, 1) void lstm_persistent(
    const float* __restrict__ x,
    const unsigned int* __restrict__ wx0, const unsigned int* __restrict__ wh0,
    const unsigned int* __restrict__ wx1, const unsigned int* __restrict__ wh1,
    const float* __restrict__ bias0, const float* __restrict__ bias1,
    float* __restrict__ h1out) {
  __shared__ __align__(16) unsigned short h0s[2][16][HPAD];
  __shared__ __align__(16) unsigned short h1s[2][16][HPAD];
  __shared__ __align__(16) unsigned short xsb[2][16][32];

  const int tid  = threadIdx.x;
  const int lane = tid & 31;
  const int wid  = tid >> 5;               // 0..7
  const int wg   = blockIdx.x;             // batch tile
  const int hi   = (lane >> 4) & 1;        // C/D layout: rows M = v + 8*hi
  const int nm   = (wid == 0) ? 3 : 2;     // number of owned m tiles

  for (int i = tid; i < 2 * 16 * HPAD; i += 256) { (&h0s[0][0][0])[i] = 0; (&h1s[0][0][0])[i] = 0; }
  for (int i = tid; i < 2 * 16 * 32;   i += 256) (&xsb[0][0][0])[i] = 0;

  float c0r[3][8], c1r[3][8];
#pragma unroll
  for (int i = 0; i < 3; i++)
#pragma unroll
    for (int v = 0; v < 8; v++) { c0r[i][v] = 0.f; c1r[i][v] = 0.f; }

  float b0r[3][4], b1r[3][4];
#pragma unroll
  for (int i = 0; i < 3; i++) {
    int m = wid + 8 * i;
    int jj = ((m < NTCH) ? m : 0) * 16 + (lane & 15);
#pragma unroll
    for (int g = 0; g < 4; g++) { b0r[i][g] = bias0[g * CHPAD + jj]; b1r[i][g] = bias1[g * CHPAD + jj]; }
  }

  // stage x_0 into buffer 0
  if (tid < 16 * INS) {
    int b = tid / INS, k = tid - b * INS;
    xsb[0][b][k] = (unsigned short)f2bf(x[((size_t)(wg * 16 + b) * TSTEPS + 0) * INS + k]);
  }
  __syncthreads();

  int p = 0;
#pragma unroll 1
  for (int t = 0; t < TSTEPS; t++) {
    const int q = p ^ 1;
    const unsigned short* h0rd = &h0s[p][0][0];
    unsigned short*       h0wr = &h0s[q][0][0];
    const unsigned short* h1rd = &h1s[p][0][0];
    unsigned short*       h1wr = &h1s[q][0][0];
    const unsigned short* xsrd = &xsb[p][0][0];
    unsigned short*       xswr = &xsb[q][0][0];

    // ---------------- layer 0: gates = x@Wx0^T + h0@Wh0^T + b ----------------
#pragma unroll
    for (int i = 0; i < 3; i++) {
      if (i < nm) {
        const int m  = wid + 8 * i;
        const int jj = m * 16 + (lane & 15);
        v8f acc[4];
#pragma unroll
        for (int g = 0; g < 4; g++) {
          v8f a;
#pragma unroll
          for (int v = 0; v < 8; v++) a[v] = b0r[i][g];
          acc[g] = a;
        }
        {
          v16bf a = load_a(xsrd, 32, lane, 0);
#pragma unroll
          for (int g = 0; g < 4; g++) acc[g] = wmma_bf16(a, load_b(wx0, g * NTCH + m, 1, 0, lane), acc[g]);
        }
        gemm_kloop(h0rd, wh0, m, lane, acc);
#pragma unroll
        for (int v = 0; v < 8; v++) {
          float ig = sigf(acc[0][v]);
          float fg = sigf(acc[1][v]);
          float gg = tanhfast(acc[2][v]);
          float og = sigf(acc[3][v]);
          float c = c0r[i][v] * fg + ig * gg;
          c0r[i][v] = c;
          h0wr[(v + hi * 8) * HPAD + jj] = (unsigned short)f2bf(og * tanhfast(c));
        }
      }
    }
    __syncthreads();   // new h0 (buffer q) visible to all waves

    // stage x_{t+1} into the other x buffer while layer 1 runs
    if (t + 1 < TSTEPS && tid < 16 * INS) {
      int b = tid / INS, k = tid - b * INS;
      xswr[b * 32 + k] = (unsigned short)f2bf(x[((size_t)(wg * 16 + b) * TSTEPS + (t + 1)) * INS + k]);
    }

    // ---------------- layer 1: gates = h0_new@Wx1^T + h1@Wh1^T + b ----------------
#pragma unroll
    for (int i = 0; i < 3; i++) {
      if (i < nm) {
        const int m  = wid + 8 * i;
        const int jj = m * 16 + (lane & 15);
        v8f acc[4];
#pragma unroll
        for (int g = 0; g < 4; g++) {
          v8f a;
#pragma unroll
          for (int v = 0; v < 8; v++) a[v] = b1r[i][g];
          acc[g] = a;
        }
        gemm_kloop(h0wr, wx1, m, lane, acc);
        gemm_kloop(h1rd, wh1, m, lane, acc);
#pragma unroll
        for (int v = 0; v < 8; v++) {
          float ig = sigf(acc[0][v]);
          float fg = sigf(acc[1][v]);
          float gg = tanhfast(acc[2][v]);
          float og = sigf(acc[3][v]);
          float c = c1r[i][v] * fg + ig * gg;
          c1r[i][v] = c;
          float hv = og * tanhfast(c);
          h1wr[(v + hi * 8) * HPAD + jj] = (unsigned short)f2bf(hv);
          if (t == TSTEPS - 1 && jj < H)
            h1out[(size_t)(wg * 16 + v + hi * 8) * H + jj] = hv;   // final state in f32
        }
      }
    }
    __syncthreads();   // h1 (buffer q) + x_{t+1} visible
    p = q;
  }
}

// ---------------- tiny FC head: 265 -> 132 -> 66 -> 1 ----------------
__global__ void fc_head(const float* __restrict__ h1,
                        const float* __restrict__ w1, const float* __restrict__ b1,
                        const float* __restrict__ w2, const float* __restrict__ b2,
                        const float* __restrict__ w3, const float* __restrict__ b3,
                        float* __restrict__ out) {
  __shared__ float hrow[H];
  __shared__ float t1[132];
  __shared__ float t2[66];
  int row = blockIdx.x, tid = threadIdx.x;
  for (int i = tid; i < H; i += blockDim.x) hrow[i] = h1[(size_t)row * H + i];
  __syncthreads();
  if (tid < 132) {
    float s = b1[tid];
    for (int k = 0; k < H; k++) s += w1[(size_t)tid * H + k] * hrow[k];
    t1[tid] = s;
  }
  __syncthreads();
  if (tid < 66) {
    float s = b2[tid];
    for (int k = 0; k < 132; k++) s += w2[(size_t)tid * 132 + k] * t1[k];
    t2[tid] = s;
  }
  __syncthreads();
  if (tid == 0) {
    float s = b3[0];
    for (int k = 0; k < 66; k++) s += w3[k] * t2[k];
    out[row] = s;
  }
}

extern "C" void kernel_launch(void* const* d_in, const int* in_sizes, int n_in,
                              void* d_out, int out_size, void* d_ws, size_t ws_size,
                              hipStream_t stream) {
  const float* x     = (const float*)d_in[0];
  const float* w_xh0 = (const float*)d_in[1];
  const float* b_xh0 = (const float*)d_in[2];
  const float* w_hh0 = (const float*)d_in[3];
  const float* b_hh0 = (const float*)d_in[4];
  const float* w_xh1 = (const float*)d_in[5];
  const float* b_xh1 = (const float*)d_in[6];
  const float* w_hh1 = (const float*)d_in[7];
  const float* b_hh1 = (const float*)d_in[8];
  const float* fc1w  = (const float*)d_in[9];
  const float* fc1b  = (const float*)d_in[10];
  const float* fc2w  = (const float*)d_in[11];
  const float* fc2b  = (const float*)d_in[12];
  const float* fc3w  = (const float*)d_in[13];
  const float* fc3b  = (const float*)d_in[14];

  // workspace layout (dwords): packed weights + combined biases + final h1
  unsigned int* ws     = (unsigned int*)d_ws;
  unsigned int* pk_wx0 = ws;                                 // 68*1*256
  unsigned int* pk_wh0 = pk_wx0 + (size_t)NTILES * 1 * 256;  // 68*9*256
  unsigned int* pk_wx1 = pk_wh0 + (size_t)NTILES * 9 * 256;
  unsigned int* pk_wh1 = pk_wx1 + (size_t)NTILES * 9 * 256;
  float* bias0 = (float*)(pk_wh1 + (size_t)NTILES * 9 * 256);
  float* bias1 = bias0 + 4 * CHPAD;
  float* h1f   = bias1 + 4 * CHPAD;                          // 512*265 f32

  pack_w<<<NTILES * 1, 256, 0, stream>>>(w_xh0, INS, 1, pk_wx0);
  pack_w<<<NTILES * 9, 256, 0, stream>>>(w_hh0, H, 9, pk_wh0);
  pack_w<<<NTILES * 9, 256, 0, stream>>>(w_xh1, H, 9, pk_wx1);
  pack_w<<<NTILES * 9, 256, 0, stream>>>(w_hh1, H, 9, pk_wh1);
  pack_bias<<<5, 256, 0, stream>>>(b_xh0, b_hh0, bias0);
  pack_bias<<<5, 256, 0, stream>>>(b_xh1, b_hh1, bias1);

  lstm_persistent<<<BATCH / 16, 256, 0, stream>>>(x, pk_wx0, pk_wh0, pk_wx1, pk_wh1,
                                                  bias0, bias1, h1f);
  fc_head<<<BATCH, 256, 0, stream>>>(h1f, fc1w, fc1b, fc2w, fc2b, fc3w, fc3b, (float*)d_out);
}
